// GRUAutoencoder_67645734912183
// MI455X (gfx1250) — matrix operand order repaired and checked by reading, not compiled, not run on hardware
//
#include <hip/hip_runtime.h>
#include <hip/hip_bf16.h>

// Problem constants (must match reference)
#define B_   256
#define TX_  50
#define TY_  100
#define NX_  64
#define NY_  64
#define H_   512
#define HOR_ 60
#define M1_  1024
#define M2_  512

typedef __attribute__((ext_vector_type(16))) _Float16 v16h;
typedef __attribute__((ext_vector_type(8)))  _Float16 v8h;
typedef __attribute__((ext_vector_type(8)))  float    v8f;

union HF16 { v16h v; v8h h[2]; };

#define TP 40   // padded LDS tile row length in halves (32 data + 8 pad -> conflict-free)

__device__ __forceinline__ float sigm_(float x) { return 1.0f / (1.0f + __expf(-x)); }
__device__ __forceinline__ float tanh_(float x) { float e = __expf(-2.0f * x); return (1.0f - e) / (1.0f + e); }

// ---------------------------------------------------------------------------
// f32 -> f16 conversion
// ---------------------------------------------------------------------------
__global__ void cvt_f16_k(const float* __restrict__ s, _Float16* __restrict__ d, long n)
{
    long i = (long)blockIdx.x * blockDim.x + threadIdx.x;
    if (i < n) d[i] = (_Float16)s[i];
}

// ---------------------------------------------------------------------------
// h_x = a + b, written as f16 into up to two strided destinations
// ---------------------------------------------------------------------------
__global__ void addpair_k(const float* __restrict__ a, const float* __restrict__ b,
                          _Float16* __restrict__ d1, long ld1,
                          _Float16* __restrict__ d2, long ld2, int cols, long n)
{
    long i = (long)blockIdx.x * blockDim.x + threadIdx.x;
    if (i >= n) return;
    long r = i / cols, c = i % cols;
    float v = a[i] + b[i];
    d1[r * ld1 + c] = (_Float16)v;
    if (d2) d2[r * ld2 + c] = (_Float16)v;
}

// ---------------------------------------------------------------------------
// Fragment loaders.
// A fragment (16x32 f16): lanes 0-15 hold K 0-7 & 16-23 of row M=lane,
//                         lanes 16-31 hold K 8-15 & 24-31 (ISA 7.12.2).
// B fragment (32x16 f16): column per lane, same K split as A.
// ---------------------------------------------------------------------------
__device__ __forceinline__ HF16 load_a_frag(const _Float16* As, int wm, int lane)
{
    HF16 af;
    const int mloc = (wm << 4) + (lane & 15);
    const int kb   = (lane >> 4) << 3;
    af.h[0] = *(const v8h*)&As[mloc * TP + kb];
    af.h[1] = *(const v8h*)&As[mloc * TP + kb + 16];
    return af;
}

__device__ __forceinline__ HF16 load_b_frag(const _Float16* Ws, int nt, int lane)
{
    HF16 bf;
    const int nloc = (nt << 4) + (lane & 15);
    const int kb   = (lane >> 4) << 3;
    bf.h[0] = *(const v8h*)&Ws[nloc * TP + kb];
    bf.h[1] = *(const v8h*)&Ws[nloc * TP + kb + 16];
    return bf;
}

// ---------------------------------------------------------------------------
// Generic WMMA GEMM: C[M,N] = A[M,K] (f16, row-major, lda) * W[N,K]^T (f16, ldw)
//                    + bias[N]; optional relu; f32 and/or f16 outputs.
// Block = 128 threads (4 waves), each block computes a 64x64 tile;
// each wave computes a 16x64 strip as 4 WMMA 16x16 accumulators.
// All B fragments are preloaded per K-step so the 4 v_wmma issue back-to-back
// behind a single LDS wait instead of one wait per WMMA.
// M, N multiples of 64; K multiple of 32 (true for all uses here).
// ---------------------------------------------------------------------------
__global__ __launch_bounds__(128)
void wgemm_k(const _Float16* __restrict__ A, long lda,
             const _Float16* __restrict__ W, long ldw,
             const float* __restrict__ bias,
             int K, int relu,
             float* __restrict__ out32, long ldc32,
             _Float16* __restrict__ out16, long ldc16)
{
    __shared__ _Float16 As[64 * TP];
    __shared__ _Float16 Ws[64 * TP];
    const int  tid  = threadIdx.x;
    const int  wm   = tid >> 5;
    const int  lane = tid & 31;
    const long rb   = (long)blockIdx.x * 64;
    const long cb   = (long)blockIdx.y * 64;

    v8f acc[4];
#pragma unroll
    for (int i = 0; i < 4; ++i) acc[i] = v8f{};

    for (int k0 = 0; k0 < K; k0 += 32) {
        __syncthreads();
        for (int i = tid; i < 256; i += 128) {       // 64 rows x 4 vec8
            int r = i >> 2, c = (i & 3) << 3;
            *(v8h*)&As[r * TP + c] = *(const v8h*)&A[(rb + r) * lda + k0 + c];
            *(v8h*)&Ws[r * TP + c] = *(const v8h*)&W[(cb + r) * ldw + k0 + c];
        }
        if (k0 + 32 < K) {                            // prefetch next K tile
            __builtin_prefetch(&A[(rb + (tid >> 1)) * lda + k0 + 32], 0, 1);
            __builtin_prefetch(&W[(cb + (tid >> 1)) * ldw + k0 + 32], 0, 1);
        }
        __syncthreads();

        HF16 af = load_a_frag(As, wm, lane);
        HF16 bf[4];
#pragma unroll
        for (int nt = 0; nt < 4; ++nt) bf[nt] = load_b_frag(Ws, nt, lane);
#pragma unroll
        for (int nt = 0; nt < 4; ++nt)
            acc[nt] = __builtin_amdgcn_wmma_f32_16x16x32_f16(
                false, af.v, false, bf[nt].v, (short)0, acc[nt], false, false);
    }

#pragma unroll
    for (int nt = 0; nt < 4; ++nt) {
#pragma unroll
        for (int v = 0; v < 8; ++v) {
            long m = rb + (wm << 4) + v + ((lane >> 4) << 3);
            long n = cb + (nt << 4) + (lane & 15);
            float val = acc[nt][v] + bias[n];
            if (relu) val = fmaxf(val, 0.0f);
            if (out32) out32[m * ldc32 + n] = val;
            if (out16) out16[m * ldc16 + n] = (_Float16)val;
        }
    }
}

// ---------------------------------------------------------------------------
// Fused GRU step: for all 3 gates g in {r,z,n}:
//   hg[g] = h @ Whh[g]^T  (+ optional x-side GEMM A2 @ W2[g]^T, decoder only)
// then   r = sig(xr + hr + bhh_r), z = sig(xz + hz + bhh_z)
//        n = tanh(xn + xacc_n + r*(hn + bhh_n))
//        h' = (1-z)*n + z*h_prev        (written as f32 and f16)
// add16 holds the precomputed x-side gate input (bih folded in), layout
// [row-stride addStride][+addOff][gate*H + n]. The decoder's y_p@Wy^T n-gate
// contribution goes into separate accumulators (accx) so r doesn't scale it.
// Per gate: 4 B fragments preloaded, then 4 back-to-back v_wmma.
// Grid (B/64, H/64), block 128.
// ---------------------------------------------------------------------------
__global__ __launch_bounds__(128)
void gru_step_k(const _Float16* __restrict__ h16, const float* __restrict__ h32,
                const _Float16* __restrict__ Whh, long ldwh,
                const _Float16* __restrict__ A2, long lda2,
                const _Float16* __restrict__ W2, long ldw2, int K2,
                const _Float16* __restrict__ add16, long addStride, long addOff,
                const float* __restrict__ bhh,
                float* __restrict__ h32o, _Float16* __restrict__ h16o)
{
    __shared__ _Float16 As[64 * TP];
    __shared__ _Float16 Ws[3 * 64 * TP];
    const int  tid  = threadIdx.x;
    const int  wm   = tid >> 5;
    const int  lane = tid & 31;
    const long rb   = (long)blockIdx.x * 64;   // batch rows
    const long cb   = (long)blockIdx.y * 64;   // gate-local column base

    v8f acc[3][4], accx[4];
#pragma unroll
    for (int g = 0; g < 3; ++g)
#pragma unroll
        for (int nt = 0; nt < 4; ++nt) acc[g][nt] = v8f{};
#pragma unroll
    for (int nt = 0; nt < 4; ++nt) accx[nt] = v8f{};

    for (int phase = 0; phase < 2; ++phase) {
        const _Float16* Ap  = phase ? A2 : h16;
        const long      lda = phase ? lda2 : (long)H_;
        const _Float16* Wp  = phase ? W2 : Whh;
        const long      ldw = phase ? ldw2 : ldwh;
        const int       Kt  = phase ? K2 : H_;
        if (Kt == 0) break;                      // encoder: no phase 1

        for (int k0 = 0; k0 < Kt; k0 += 32) {
            __syncthreads();
            for (int i = tid; i < 256; i += 128) {
                int r = i >> 2, c = (i & 3) << 3;
                *(v8h*)&As[r * TP + c] = *(const v8h*)&Ap[(rb + r) * lda + k0 + c];
            }
            for (int g = 0; g < 3; ++g) {
                for (int i = tid; i < 256; i += 128) {
                    int r = i >> 2, c = (i & 3) << 3;
                    *(v8h*)&Ws[g * 64 * TP + r * TP + c] =
                        *(const v8h*)&Wp[((long)g * H_ + cb + r) * ldw + k0 + c];
                }
            }
            if (k0 + 32 < Kt)
                __builtin_prefetch(&Wp[(cb + (tid >> 1)) * ldw + k0 + 32], 0, 1);
            __syncthreads();

            HF16 af = load_a_frag(As, wm, lane);
#pragma unroll
            for (int g = 0; g < 3; ++g) {
                const _Float16* Wsg = Ws + g * 64 * TP;
                HF16 bf[4];
#pragma unroll
                for (int nt = 0; nt < 4; ++nt) bf[nt] = load_b_frag(Wsg, nt, lane);
#pragma unroll
                for (int nt = 0; nt < 4; ++nt) {
                    v8f& tgt = (phase == 1 && g == 2) ? accx[nt] : acc[g][nt];
                    tgt = __builtin_amdgcn_wmma_f32_16x16x32_f16(
                        false, af.v, false, bf[nt].v, (short)0, tgt, false, false);
                }
            }
        }
    }

#pragma unroll
    for (int nt = 0; nt < 4; ++nt) {
#pragma unroll
        for (int v = 0; v < 8; ++v) {
            long m = rb + (wm << 4) + v + ((lane >> 4) << 3);
            long n = cb + (nt << 4) + (lane & 15);
            const _Float16* ap = add16 + m * addStride + addOff;
            float xr = (float)ap[0 * H_ + n];
            float xz = (float)ap[1 * H_ + n];
            float xn = (float)ap[2 * H_ + n] + accx[nt][v];
            float hr = acc[0][nt][v] + bhh[0 * H_ + n];
            float hz = acc[1][nt][v] + bhh[1 * H_ + n];
            float hn = acc[2][nt][v] + bhh[2 * H_ + n];
            float r  = sigm_(xr + hr);
            float z  = sigm_(xz + hz);
            float nn = tanh_(xn + r * hn);
            float hp = h32[m * H_ + n];
            float ho = (1.0f - z) * nn + z * hp;
            h32o[m * H_ + n] = ho;
            h16o[m * H_ + n] = (_Float16)ho;
        }
    }
}

// ---------------------------------------------------------------------------
// Host side
// ---------------------------------------------------------------------------
extern "C" void kernel_launch(void* const* d_in, const int* in_sizes, int n_in,
                              void* d_out, int out_size, void* d_ws, size_t ws_size,
                              hipStream_t stream)
{
    (void)in_sizes; (void)n_in; (void)out_size; (void)ws_size;

    // Input order per setup_inputs()
    const float* xf_bih = (const float*)d_in[4];
    const float* xf_bhh = (const float*)d_in[5];
    const float* xb_bih = (const float*)d_in[8];
    const float* xb_bhh = (const float*)d_in[9];
    const float* ef_bih = (const float*)d_in[12];
    const float* ef_bhh = (const float*)d_in[13];
    const float* eb_bih = (const float*)d_in[16];
    const float* eb_bhh = (const float*)d_in[17];
    const float* em_b1  = (const float*)d_in[19];
    const float* em_b2  = (const float*)d_in[21];
    const float* eo_b   = (const float*)d_in[23];
    const float* d_bih  = (const float*)d_in[26];
    const float* d_bhh  = (const float*)d_in[27];
    const float* dm_b1  = (const float*)d_in[29];
    const float* dm_b2  = (const float*)d_in[31];
    const float* do_b   = (const float*)d_in[33];

    // Workspace carve (256B aligned)
    char* p = (char*)d_ws;
    auto carve = [&](size_t bytes) -> void* {
        void* r = (void*)p;
        p += (bytes + 255) & ~(size_t)255;
        return r;
    };

    _Float16* x16    = (_Float16*)carve((size_t)B_ * TX_ * NX_ * 2);
    _Float16* y16    = (_Float16*)carve((size_t)B_ * TY_ * NY_ * 2);
    _Float16* xfWih16 = (_Float16*)carve((size_t)3 * H_ * NX_ * 2);
    _Float16* xfWhh16 = (_Float16*)carve((size_t)3 * H_ * H_ * 2);
    _Float16* xbWih16 = (_Float16*)carve((size_t)3 * H_ * NX_ * 2);
    _Float16* xbWhh16 = (_Float16*)carve((size_t)3 * H_ * H_ * 2);
    _Float16* efWih16 = (_Float16*)carve((size_t)3 * H_ * NY_ * 2);
    _Float16* efWhh16 = (_Float16*)carve((size_t)3 * H_ * H_ * 2);
    _Float16* ebWih16 = (_Float16*)carve((size_t)3 * H_ * NY_ * 2);
    _Float16* ebWhh16 = (_Float16*)carve((size_t)3 * H_ * H_ * 2);
    _Float16* emW116  = (_Float16*)carve((size_t)M1_ * 2 * H_ * 2);
    _Float16* emW216  = (_Float16*)carve((size_t)M2_ * M1_ * 2);
    _Float16* eoW16   = (_Float16*)carve((size_t)H_ * M2_ * 2);
    _Float16* dWih16  = (_Float16*)carve((size_t)3 * H_ * (2 * H_ + NY_) * 2);
    _Float16* dWhh16  = (_Float16*)carve((size_t)3 * H_ * H_ * 2);
    _Float16* dmW116  = (_Float16*)carve((size_t)M1_ * H_ * 2);
    _Float16* dmW216  = (_Float16*)carve((size_t)M2_ * M1_ * 2);
    _Float16* doW16   = (_Float16*)carve((size_t)NY_ * M2_ * 2);
    _Float16* xg16    = (_Float16*)carve((size_t)B_ * TY_ * 3 * H_ * 2);  // shared across GRU runs
    _Float16* hA16[2]; float* hA32[2];
    _Float16* hB16[2]; float* hB32[2];
    for (int i = 0; i < 2; ++i) {
        hA16[i] = (_Float16*)carve((size_t)B_ * H_ * 2);
        hA32[i] = (float*)carve((size_t)B_ * H_ * 4);
        hB16[i] = (_Float16*)carve((size_t)B_ * H_ * 2);
        hB32[i] = (float*)carve((size_t)B_ * H_ * 4);
    }
    _Float16* hcat16 = (_Float16*)carve((size_t)B_ * 2 * H_ * 2);  // [h_x | h_y]
    _Float16* hxz16  = (_Float16*)carve((size_t)B_ * 2 * H_ * 2);  // [h_x | z]
    _Float16* m1b    = (_Float16*)carve((size_t)B_ * M1_ * 2);
    _Float16* m2b    = (_Float16*)carve((size_t)B_ * M2_ * 2);
    _Float16* cst16  = (_Float16*)carve((size_t)B_ * 3 * H_ * 2);
    _Float16* yp16[2];
    yp16[0] = (_Float16*)carve((size_t)B_ * NY_ * 2);
    yp16[1] = (_Float16*)carve((size_t)B_ * NY_ * 2);

    auto cvt = [&](const void* s, _Float16* d, long n) {
        cvt_f16_k<<<dim3((unsigned)((n + 255) / 256)), 256, 0, stream>>>((const float*)s, d, n);
    };
    cvt(d_in[0],  x16,     (long)B_ * TX_ * NX_);
    cvt(d_in[1],  y16,     (long)B_ * TY_ * NY_);
    cvt(d_in[2],  xfWih16, (long)3 * H_ * NX_);
    cvt(d_in[3],  xfWhh16, (long)3 * H_ * H_);
    cvt(d_in[6],  xbWih16, (long)3 * H_ * NX_);
    cvt(d_in[7],  xbWhh16, (long)3 * H_ * H_);
    cvt(d_in[10], efWih16, (long)3 * H_ * NY_);
    cvt(d_in[11], efWhh16, (long)3 * H_ * H_);
    cvt(d_in[14], ebWih16, (long)3 * H_ * NY_);
    cvt(d_in[15], ebWhh16, (long)3 * H_ * H_);
    cvt(d_in[18], emW116,  (long)M1_ * 2 * H_);
    cvt(d_in[20], emW216,  (long)M2_ * M1_);
    cvt(d_in[22], eoW16,   (long)H_ * M2_);
    cvt(d_in[24], dWih16,  (long)3 * H_ * (2 * H_ + NY_));
    cvt(d_in[25], dWhh16,  (long)3 * H_ * H_);
    cvt(d_in[28], dmW116,  (long)M1_ * H_);
    cvt(d_in[30], dmW216,  (long)M2_ * M1_);
    cvt(d_in[32], doW16,   (long)NY_ * M2_);

    auto gemm = [&](const _Float16* A, long lda, const _Float16* W, long ldw,
                    const float* bias, int M, int N, int K, int relu,
                    float* o32, long l32, _Float16* o16, long l16) {
        wgemm_k<<<dim3((unsigned)(M / 64), (unsigned)(N / 64)), 128, 0, stream>>>(
            A, lda, W, ldw, bias, K, relu, o32, l32, o16, l16);
    };

    auto grurun = [&](const _Float16* in16, int T, int Nin,
                      const _Float16* Wih16, const _Float16* Whh16,
                      const float* bih, const float* bhh,
                      _Float16** h16p, float** h32p, bool rev) {
        // All-timestep input projection: xg[b][t][g] = x @ Wih^T + bih   (f16)
        gemm(in16, Nin, Wih16, Nin, bih, B_ * T, 3 * H_, Nin, 0,
             nullptr, 0, xg16, 3 * H_);
        hipMemsetAsync(h16p[0], 0, (size_t)B_ * H_ * 2, stream);
        hipMemsetAsync(h32p[0], 0, (size_t)B_ * H_ * 4, stream);
        for (int s = 0; s < T; ++s) {
            int t = rev ? (T - 1 - s) : s;
            gru_step_k<<<dim3(B_ / 64, H_ / 64), 128, 0, stream>>>(
                h16p[s & 1], h32p[s & 1], Whh16, (long)H_,
                nullptr, 0, nullptr, 0, 0,
                xg16, (long)T * 3 * H_, (long)t * 3 * H_, bhh,
                h32p[(s + 1) & 1], h16p[(s + 1) & 1]);
        }
    };

    const long nh = (long)B_ * H_;
    const int fx = TX_ & 1, fy = TY_ & 1;   // final h ping-pong index

    // ---- Encoder GRUs over x ----
    grurun(x16, TX_, NX_, xfWih16, xfWhh16, xf_bih, xf_bhh, hA16, hA32, false);
    grurun(x16, TX_, NX_, xbWih16, xbWhh16, xb_bih, xb_bhh, hB16, hB32, true);
    // h_x = hT_f + hT_b -> hcat[:, :H] and hxz[:, :H]
    addpair_k<<<dim3((unsigned)((nh + 255) / 256)), 256, 0, stream>>>(
        hA32[fx], hB32[fx], hcat16, 2 * H_, hxz16, 2 * H_, H_, nh);

    // ---- Encoder GRUs over y ----
    grurun(y16, TY_, NY_, efWih16, efWhh16, ef_bih, ef_bhh, hA16, hA32, false);
    grurun(y16, TY_, NY_, ebWih16, ebWhh16, eb_bih, eb_bhh, hB16, hB32, true);
    addpair_k<<<dim3((unsigned)((nh + 255) / 256)), 256, 0, stream>>>(
        hA32[fy], hB32[fy], hcat16 + H_, 2 * H_, nullptr, 0, H_, nh);

    // ---- Encoder MLP: h -> relu -> relu -> z ----
    gemm(hcat16, 2 * H_, emW116, 2 * H_, em_b1, B_, M1_, 2 * H_, 1, nullptr, 0, m1b, M1_);
    gemm(m1b, M1_, emW216, M1_, em_b2, B_, M2_, M1_, 1, nullptr, 0, m2b, M2_);
    gemm(m2b, M2_, eoW16, M2_, eo_b, B_, H_, M2_, 0, nullptr, 0, hxz16 + H_, 2 * H_);

    // ---- const = [h_x | z] @ d_Wih[:, :2H]^T + d_bih ----
    gemm(hxz16, 2 * H_, dWih16, 2 * H_ + NY_, d_bih, B_, 3 * H_, 2 * H_, 0,
         nullptr, 0, cst16, 3 * H_);

    // ---- Decoder ----
    hipMemsetAsync(hA16[0], 0, (size_t)B_ * H_ * 2, stream);
    hipMemsetAsync(hA32[0], 0, (size_t)B_ * H_ * 4, stream);
    float* out = (float*)d_out;
    for (int s = 0; s < HOR_; ++s) {
        const _Float16* a2 = (s == 0) ? (x16 + (size_t)(TX_ - 1) * NX_) : yp16[(s + 1) & 1];
        long lda2 = (s == 0) ? (long)TX_ * NX_ : (long)NY_;
        gru_step_k<<<dim3(B_ / 64, H_ / 64), 128, 0, stream>>>(
            hA16[s & 1], hA32[s & 1], dWhh16, (long)H_,
            a2, lda2, dWih16 + 2 * H_, (long)(2 * H_ + NY_), NY_,
            cst16, (long)3 * H_, 0, d_bhh,
            hA32[(s + 1) & 1], hA16[(s + 1) & 1]);
        _Float16* hn = hA16[(s + 1) & 1];
        gemm(hn, H_, dmW116, H_, dm_b1, B_, M1_, H_, 1, nullptr, 0, m1b, M1_);
        gemm(m1b, M1_, dmW216, M1_, dm_b2, B_, M2_, M1_, 1, nullptr, 0, m2b, M2_);
        // y_i: f32 into d_out (B, HOR, NY) and f16 into next-step y_p buffer
        gemm(m2b, M2_, doW16, M2_, do_b, B_, NY_, M2_, 0,
             out + (size_t)s * NY_, (long)HOR_ * NY_, yp16[s & 1], NY_);
    }
}
